// AdGraphAttn_42984032698882
// MI455X (gfx1250) — compile-verified, matrix operand order checked by reading, not run on hardware
//
#include <hip/hip_runtime.h>
#include <hip/hip_bf16.h>
#include <math.h>

// Problem constants (from reference)
#define NN 50000
#define EE 400000
#define LL 1024
#define DD 256
#define HH 4
#define CC 64
#define MT   3125   /* N/16 row tiles  */
#define KT_L 32     /* L/32 k tiles    */
#define NT_L 64     /* 4D/16 col tiles */
#define KT_P 8      /* D/32 k tiles    */
#define NT_P 16     /* D/16 col tiles  */

typedef __attribute__((ext_vector_type(16))) __bf16 v16bf;
typedef __attribute__((ext_vector_type(8)))  float  v8f;

static __device__ __forceinline__ __bf16 f2b(float f) { return (__bf16)f; }

// K-offset inside a 16-bit WMMA fragment (ISA 7.12.2, 16-bit A 16x32 layout):
// lane half (0: K 0..7 / 16..23, 1: K 8..15 / 24..31), elem pairs per VGPR.
static __device__ __forceinline__ int frag_k(int j, int half) {
  int pair = j >> 1;
  int base = (pair < 4) ? (pair * 2) : (16 + (pair - 4) * 2);
  return base + half * 8 + (j & 1);
}

static __device__ __forceinline__ void atomicMaxF(float* addr, float val) {
  int* ai = (int*)addr;
  int cur = __float_as_int(*addr);
  while (__int_as_float(cur) < val) {
    int old = atomicCAS(ai, cur, __float_as_int(val));
    if (old == cur) break;
    cur = old;
  }
}

// ---------------- fill ----------------
__global__ void fill_kernel(float* __restrict__ p, size_t n, float v) {
  size_t i = (size_t)blockIdx.x * 256 + threadIdx.x;
  if (i < n) p[i] = v;
}

// ---------------- packing to WMMA fragment layout ----------------
// A-pack layout: [((mb*Kt + kb)*512) + lane*16 + j] bf16
__global__ void pack_a_kernel(const float* __restrict__ X, __bf16* __restrict__ Apack,
                              int Kt, int K, int Nrows, size_t total) {
  size_t idx = (size_t)blockIdx.x * 256 + threadIdx.x;
  if (idx >= total) return;
  int j = (int)(idx & 15);
  int l = (int)((idx >> 4) & 31);
  size_t tile = idx >> 9;
  int kb = (int)(tile % Kt);
  int mb = (int)(tile / Kt);
  int m = mb * 16 + (l & 15);
  int k = kb * 32 + frag_k(j, l >> 4);
  float v = (m < Nrows) ? X[(size_t)m * K + k] : 0.0f;
  Apack[idx] = f2b(v);
}

// B-pack layout: [((kb*Ntiles + nb)*512) + lane*16 + j] bf16
__global__ void pack_b_kernel(const float* __restrict__ W, __bf16* __restrict__ Bpack,
                              int Ntiles, int Ncols, size_t total) {
  size_t idx = (size_t)blockIdx.x * 256 + threadIdx.x;
  if (idx >= total) return;
  int j = (int)(idx & 15);
  int l = (int)((idx >> 4) & 31);
  size_t tile = idx >> 9;
  int nb = (int)(tile % Ntiles);
  int kb = (int)(tile / Ntiles);
  int k = kb * 32 + frag_k(j, l >> 4);
  int n = nb * 16 + (l & 15);
  Bpack[idx] = f2b(W[(size_t)k * Ncols + n]);
}

// Pack concatenated [Wq|Wk|Wv|Wskip] (each L x D) as one L x 4D B operand.
__global__ void pack_b4_kernel(const float* __restrict__ Wq, const float* __restrict__ Wk,
                               const float* __restrict__ Wv, const float* __restrict__ Ws,
                               __bf16* __restrict__ Bpack, size_t total) {
  size_t idx = (size_t)blockIdx.x * 256 + threadIdx.x;
  if (idx >= total) return;
  int j = (int)(idx & 15);
  int l = (int)((idx >> 4) & 31);
  size_t tile = idx >> 9;
  int nb = (int)(tile % NT_L);
  int kb = (int)(tile / NT_L);
  int k = kb * 32 + frag_k(j, l >> 4);
  int n = nb * 16 + (l & 15);          // 0..1023 across the four matrices
  int sel = n >> 8;
  int col = n & 255;
  const float* W = (sel == 0) ? Wq : (sel == 1) ? Wk : (sel == 2) ? Wv : Ws;
  Bpack[idx] = f2b(W[(size_t)k * DD + col]);
}

__global__ void bias4_kernel(const float* __restrict__ bq, const float* __restrict__ bk,
                             const float* __restrict__ bv, const float* __restrict__ bs,
                             float* __restrict__ bias4) {
  int n = blockIdx.x * 256 + threadIdx.x;
  if (n >= 4 * DD) return;
  int sel = n >> 8, col = n & 255;
  const float* b = (sel == 0) ? bq : (sel == 1) ? bk : (sel == 2) ? bv : bs;
  bias4[n] = b[col];
}

// ---------------- tiled bf16 WMMA GEMM ----------------
// Block = 8 waves; wave w computes row tile (blockIdx.x*8+w), 8 col tiles
// (blockIdx.y*8 .. +7). B fragments staged via LDS, shared by all waves.
__global__ __launch_bounds__(256)
void gemm_bf16_kernel(const __bf16* __restrict__ Apack, const __bf16* __restrict__ Bpack,
                      const float* __restrict__ bias, float* __restrict__ Out,
                      int Mtiles, int Kt, int Ntiles, int Ncols, int Nrows) {
  __shared__ __align__(32) __bf16 lbs[8 * 512];
  const int tid  = threadIdx.x;
  const int wave = tid >> 5;
  const int lane = tid & 31;
  const int mb   = blockIdx.x * 8 + wave;
  const int nb0  = blockIdx.y * 8;
  const bool mvalid = (mb < Mtiles);

  v8f acc[8];
  v8f zero = {};
#pragma unroll
  for (int t = 0; t < 8; ++t) acc[t] = zero;

  const __bf16* aptr = Apack + ((size_t)mb * Kt) * 512 + lane * 16;

  for (int kb = 0; kb < Kt; ++kb) {
    // stage the 8 shared B fragments for this k-slice (8 KB)
    {
      const __bf16* src = Bpack + ((size_t)(kb * Ntiles + nb0 + wave)) * 512 + lane * 16;
      *(v16bf*)(&lbs[wave * 512 + lane * 16]) = *(const v16bf*)src;
    }
    __syncthreads();

    v16bf a = {};
    if (mvalid) {
      a = *(const v16bf*)aptr;
      __builtin_prefetch(aptr + 512, 0, 1);   // global_prefetch_b8 next A frag
    }
    aptr += 512;

#pragma unroll
    for (int t = 0; t < 8; ++t) {
      v16bf b = *(const v16bf*)(&lbs[t * 512 + lane * 16]);
      acc[t] = __builtin_amdgcn_wmma_f32_16x16x32_bf16(
          /*neg_a=*/false, a, /*neg_b=*/false, b,
          /*c_mod=*/(short)0, acc[t], /*reuse_a=*/false, /*reuse_b=*/false);
    }
    __syncthreads();
  }

  if (!mvalid) return;
  // C/D layout: VGPR r -> row r (lanes 0-15) / row 8+r (lanes 16-31); col = lane%16
  const int colb  = lane & 15;
  const int rhalf = (lane >> 4) * 8;
#pragma unroll
  for (int t = 0; t < 8; ++t) {
    int n = (nb0 + t) * 16 + colb;
    float bv = bias ? bias[n] : 0.0f;
#pragma unroll
    for (int r = 0; r < 8; ++r) {
      int m = mb * 16 + rhalf + r;
      if (m < Nrows) Out[(size_t)m * Ncols + n] = acc[t][r] + bv;
    }
  }
}

// ---------------- edge attention ----------------
// qkvs layout per node: [q(256) | k(256) | v(256) | skip(256)]
__global__ void edge_score_kernel(const float* __restrict__ qkvs, const int* __restrict__ ei,
                                  float* __restrict__ s, float* __restrict__ m) {
  int idx = blockIdx.x * 256 + threadIdx.x;
  if (idx >= EE * HH) return;
  int e = idx >> 2, hh = idx & 3;
  int src = ei[e], dst = ei[EE + e];
  const float* qp = qkvs + (size_t)dst * 1024 + hh * CC;          // q of dst
  const float* kp = qkvs + (size_t)src * 1024 + 256 + hh * CC;    // k of src
  float acc = 0.0f;
#pragma unroll 8
  for (int c = 0; c < CC; ++c) acc += qp[c] * kp[c];
  acc *= 0.125f;                                                   // 1/sqrt(64)
  s[idx] = acc;
  atomicMaxF(&m[dst * HH + hh], acc);
}

__global__ void edge_exp_kernel(float* __restrict__ s, const int* __restrict__ ei,
                                const float* __restrict__ m, float* __restrict__ den) {
  int idx = blockIdx.x * 256 + threadIdx.x;
  if (idx >= EE * HH) return;
  int e = idx >> 2, hh = idx & 3;
  int dst = ei[EE + e];
  float ex = expf(s[idx] - m[dst * HH + hh]);
  s[idx] = ex;
  atomicAdd(&den[dst * HH + hh], ex);
}

__global__ void edge_agg_kernel(const float* __restrict__ s, const int* __restrict__ ei,
                                const float* __restrict__ den, const float* __restrict__ qkvs,
                                float* __restrict__ agg) {
  long long idx = (long long)blockIdx.x * 256 + threadIdx.x;
  if (idx >= (long long)EE * CC) return;
  int e = (int)(idx >> 6), c = (int)(idx & 63);
  int src = ei[e], dst = ei[EE + e];
#pragma unroll
  for (int hh = 0; hh < HH; ++hh) {
    float alpha = s[e * HH + hh] / den[dst * HH + hh];
    float vv = qkvs[(size_t)src * 1024 + 512 + hh * CC + c];       // v of src
    atomicAdd(&agg[(size_t)dst * DD + hh * CC + c], alpha * vv);
  }
}

// ---------------- h = gelu(agg + skip) ----------------
__global__ void h_kernel(const float* __restrict__ agg, const float* __restrict__ qkvs,
                         float* __restrict__ hout) {
  long long idx = (long long)blockIdx.x * 256 + threadIdx.x;
  if (idx >= (long long)NN * DD) return;
  int n = (int)(idx >> 8), d = (int)(idx & 255);
  float v = agg[idx] + qkvs[(size_t)n * 1024 + 768 + d];           // skip
  hout[idx] = 0.5f * v * (1.0f + erff(v * 0.70710678118654752f));  // exact gelu
}

// ---------------- gate softmax over N ----------------
__global__ void gate_score_kernel(const float* __restrict__ h, const float* __restrict__ gw,
                                  const float* __restrict__ gb, float* __restrict__ sc,
                                  float* __restrict__ gmax) {
  int n = blockIdx.x * 256 + threadIdx.x;
  if (n >= NN) return;
  const float* hr = h + (size_t)n * DD;
  float s0 = gb[0], s1 = gb[1];
#pragma unroll 8
  for (int d = 0; d < DD; ++d) {
    float hv = hr[d];
    s0 += hv * gw[d * 2];
    s1 += hv * gw[d * 2 + 1];
  }
  sc[n * 2] = s0;
  sc[n * 2 + 1] = s1;
  atomicMaxF(&gmax[0], s0);
  atomicMaxF(&gmax[1], s1);
}

__global__ void gate_exp_kernel(float* __restrict__ sc, const float* __restrict__ gmax,
                                float* __restrict__ gsum) {
  int idx = blockIdx.x * 256 + threadIdx.x;
  if (idx >= NN * 2) return;
  float ex = expf(sc[idx] - gmax[idx & 1]);
  sc[idx] = ex;
  atomicAdd(&gsum[idx & 1], ex);
}

__global__ void attn_kernel(const float* __restrict__ sc, const float* __restrict__ gsum,
                            const int* __restrict__ label, float* __restrict__ attn,
                            float* __restrict__ A) {
  int idx = blockIdx.x * 256 + threadIdx.x;
  if (idx >= NN * 2) return;
  int c = idx & 1;
  float a = sc[idx] / gsum[c];
  attn[idx] = a;
  if (c == label[0]) A[idx >> 1] = a;
}

// ---------------- y[c][d] = sum_n attn[n][c] * p[n][d] ----------------
__global__ void pool_reduce_kernel(const float* __restrict__ p, const float* __restrict__ attn,
                                   float* __restrict__ y) {
  int d = threadIdx.x;            // 256 columns
  int row0 = blockIdx.x * 128;
  float a0 = 0.0f, a1 = 0.0f;
  for (int r = 0; r < 128; ++r) {
    int n = row0 + r;
    if (n >= NN) break;
    float pv = p[(size_t)n * DD + d];
    a0 += attn[n * 2] * pv;
    a1 += attn[n * 2 + 1] * pv;
  }
  atomicAdd(&y[d], a0);
  atomicAdd(&y[DD + d], a1);
}

// =====================================================================
extern "C" void kernel_launch(void* const* d_in, const int* in_sizes, int n_in,
                              void* d_out, int out_size, void* d_ws, size_t ws_size,
                              hipStream_t stream) {
  (void)in_sizes; (void)n_in; (void)out_size; (void)ws_size;
  const float* x      = (const float*)d_in[0];
  const int*   ei     = (const int*)  d_in[1];
  const int*   label  = (const int*)  d_in[2];
  const float* Wq     = (const float*)d_in[3];
  const float* bq     = (const float*)d_in[4];
  const float* Wk     = (const float*)d_in[5];
  const float* bk     = (const float*)d_in[6];
  const float* Wv     = (const float*)d_in[7];
  const float* bv     = (const float*)d_in[8];
  const float* Wskip  = (const float*)d_in[9];
  const float* bskip  = (const float*)d_in[10];
  const float* gate_w = (const float*)d_in[11];
  const float* gate_b = (const float*)d_in[12];
  const float* pool_w = (const float*)d_in[13];
  const float* pool_b = (const float*)d_in[14];

  // d_out regions (return order): y(2*256) | attn(N*2) | h(N*256) | A(N)
  float* y_out    = (float*)d_out;
  float* attn_out = y_out + 2 * DD;
  float* h_out    = attn_out + (size_t)NN * 2;
  float* A_out    = h_out + (size_t)NN * DD;

  // carve workspace
  char* wp = (char*)d_ws;
  auto alloc = [&](size_t bytes) { char* r = wp; wp += (bytes + 255) & ~(size_t)255; return r; };
  const size_t xpack_elems  = (size_t)MT * KT_L * 512;
  const size_t wpack_elems  = (size_t)KT_L * NT_L * 512;
  const size_t hpack_elems  = (size_t)MT * KT_P * 512;
  const size_t pwpack_elems = (size_t)KT_P * NT_P * 512;
  __bf16* xpack  = (__bf16*)alloc(xpack_elems * 2);
  __bf16* wpack  = (__bf16*)alloc(wpack_elems * 2);
  __bf16* hpack  = (__bf16*)alloc(hpack_elems * 2);
  __bf16* pwpack = (__bf16*)alloc(pwpack_elems * 2);
  float* bias4 = (float*)alloc(4 * DD * 4);
  float* qkvs  = (float*)alloc((size_t)NN * 1024 * 4);
  float* sbuf  = (float*)alloc((size_t)EE * HH * 4);
  float* mbuf  = (float*)alloc((size_t)NN * HH * 4);
  float* den   = (float*)alloc((size_t)NN * HH * 4);
  float* agg   = (float*)alloc((size_t)NN * DD * 4);
  float* sc    = (float*)alloc((size_t)NN * 2 * 4);
  float* gmax  = (float*)alloc(2 * 4);
  float* gsum  = (float*)alloc(2 * 4);
  float* pbuf  = (float*)alloc((size_t)NN * DD * 4);

  auto blk = [](size_t n) { return (unsigned)((n + 255) / 256); };

  // ---- init accumulators (d_out/d_ws are poisoned) ----
  fill_kernel<<<blk((size_t)NN * DD), 256, 0, stream>>>(agg, (size_t)NN * DD, 0.0f);
  fill_kernel<<<blk((size_t)NN * HH), 256, 0, stream>>>(mbuf, (size_t)NN * HH, -INFINITY);
  fill_kernel<<<blk((size_t)NN * HH), 256, 0, stream>>>(den, (size_t)NN * HH, 0.0f);
  fill_kernel<<<1, 256, 0, stream>>>(gmax, 2, -INFINITY);
  fill_kernel<<<1, 256, 0, stream>>>(gsum, 2, 0.0f);
  fill_kernel<<<blk(2 * DD), 256, 0, stream>>>(y_out, 2 * DD, 0.0f);

  // ---- fused QKV+skip projection: x @ [Wq|Wk|Wv|Wskip] ----
  pack_a_kernel<<<blk(xpack_elems), 256, 0, stream>>>(x, xpack, KT_L, LL, NN, xpack_elems);
  pack_b4_kernel<<<blk(wpack_elems), 256, 0, stream>>>(Wq, Wk, Wv, Wskip, wpack, wpack_elems);
  bias4_kernel<<<4, 256, 0, stream>>>(bq, bk, bv, bskip, bias4);
  {
    dim3 g((MT + 7) / 8, NT_L / 8);
    gemm_bf16_kernel<<<g, 256, 0, stream>>>(xpack, wpack, bias4, qkvs, MT, KT_L, NT_L, 1024, NN);
  }

  // ---- edge attention with segment softmax ----
  edge_score_kernel<<<blk((size_t)EE * HH), 256, 0, stream>>>(qkvs, ei, sbuf, mbuf);
  edge_exp_kernel<<<blk((size_t)EE * HH), 256, 0, stream>>>(sbuf, ei, mbuf, den);
  edge_agg_kernel<<<blk((size_t)EE * CC), 256, 0, stream>>>(sbuf, ei, den, qkvs, agg);

  // ---- h = gelu(agg + skip) (written straight to d_out) ----
  h_kernel<<<blk((size_t)NN * DD), 256, 0, stream>>>(agg, qkvs, h_out);

  // ---- gate softmax over all nodes ----
  gate_score_kernel<<<blk(NN), 256, 0, stream>>>(h_out, gate_w, gate_b, sc, gmax);
  gate_exp_kernel<<<blk((size_t)NN * 2), 256, 0, stream>>>(sc, gmax, gsum);
  attn_kernel<<<blk((size_t)NN * 2), 256, 0, stream>>>(sc, gsum, label, attn_out, A_out);

  // ---- pool GEMM: p = h @ pool_w + pool_b (WMMA again) ----
  pack_a_kernel<<<blk(hpack_elems), 256, 0, stream>>>(h_out, hpack, KT_P, DD, NN, hpack_elems);
  pack_b_kernel<<<blk(pwpack_elems), 256, 0, stream>>>(pool_w, pwpack, NT_P, DD, pwpack_elems);
  {
    dim3 g((MT + 7) / 8, NT_P / 8);
    gemm_bf16_kernel<<<g, 256, 0, stream>>>(hpack, pwpack, pool_b, pbuf, MT, KT_P, NT_P, DD, NN);
  }

  // ---- y = attn^T @ p ----
  pool_reduce_kernel<<<(NN + 127) / 128, 256, 0, stream>>>(pbuf, attn_out, y_out);
}
